// TimesBlock_19275813225118
// MI455X (gfx1250) — matrix operand order repaired and checked by reading, not compiled, not run on hardware
//
#include <hip/hip_runtime.h>
#include <math.h>

// ---------------- problem constants ----------------
#define B_   8
#define T_   2048
#define D_   64
#define K_   5
#define U_   (2*T_-2)      // 4094
#define UPAD 4096          // padded row count for h1
#define EPSB 1e-5f

typedef _Float16 half_t;
typedef __attribute__((ext_vector_type(16))) _Float16 v16h;
typedef __attribute__((ext_vector_type(8)))  _Float16 v8h;
typedef __attribute__((ext_vector_type(8)))  float    v8f;

// ---------------- workspace layout (bytes) ----------------
// s      : B*T*D f32        @ 0          (4,194,304)
// trend  : B*T*D f32        @ 4,194,304  (4,194,304)
// r      : B*T   f32        @ 8,388,608  (65,536)
// idx    : B*K   i32        @ 8,454,144  (256)
// wgt    : B*K   f32        @ 8,454,400  (256)
// wf1    : 9*2*4*32*16 f16  @ 8,454,656  (73,728)
// wf2    : same             @ 8,528,384  (73,728)
// h1     : 40*4096*64 f16   @ 8,602,112  (20,971,520)
// yp     : 40*2048*64 f32   @ 29,573,632 (20,971,520)
// s_h    : B*T*D f16        @ 50,545,152 (2,097,152)
// total ~ 52.6 MB
#define OFF_S     0
#define OFF_TREND 4194304u
#define OFF_R     8388608u
#define OFF_IDX   8454144u
#define OFF_WGT   8454400u
#define OFF_WF1   8454656u
#define OFF_WF2   8528384u
#define OFF_H1    8602112u
#define OFF_YP    29573632u
#define OFF_SH    50545152u

#define WMMA(A,Bv,C) __builtin_amdgcn_wmma_f32_16x16x32_f16(false,(A),false,(Bv),(short)0,(C),false,false)

// ============ 1) trend + seasonal (f32 + f16 copies) ============
__global__ void k_trend(const float* __restrict__ x, float* __restrict__ trend,
                        float* __restrict__ s, half_t* __restrict__ s_h) {
    int i = blockIdx.x * blockDim.x + threadIdx.x;
    if (i >= B_*T_*D_) return;
    int d = i % D_;
    int t = (i / D_) % T_;
    int b = i / (D_ * T_);
    int lo = t - 12; if (lo < 0) lo = 0;
    int hi = t + 13; if (hi > T_) hi = T_;
    const float* base = x + (size_t)b * T_ * D_ + d;
    float sum = 0.f;
    for (int tt = lo; tt < hi; ++tt) sum += base[(size_t)tt * D_];
    float tr = sum / (float)(hi - lo);
    trend[i] = tr;
    float sv = x[i] - tr;
    s[i] = sv;
    s_h[i] = (half_t)sv;
}

// ============ 2) circular autocorrelation (== irfft(|rfft|^2).mean(d)) ============
__global__ void k_autocorr(const float* __restrict__ s, float* __restrict__ r) {
    int tau = blockIdx.x;
    int b   = blockIdx.y;
    const float* sb = s + (size_t)b * T_ * D_;
    float acc = 0.f;
    for (int t = threadIdx.x; t < T_; t += 256) {
        int t2 = t + tau; if (t2 >= T_) t2 -= T_;
        const float4* p1 = reinterpret_cast<const float4*>(sb + (size_t)t  * D_);
        const float4* p2 = reinterpret_cast<const float4*>(sb + (size_t)t2 * D_);
        #pragma unroll
        for (int q = 0; q < 16; ++q) {
            float4 a = p1[q], c = p2[q];
            acc += a.x*c.x + a.y*c.y + a.z*c.z + a.w*c.w;
        }
    }
    __shared__ float red[256];
    red[threadIdx.x] = acc;
    __syncthreads();
    for (int off = 128; off > 0; off >>= 1) {
        if (threadIdx.x < (unsigned)off) red[threadIdx.x] += red[threadIdx.x + off];
        __syncthreads();
    }
    if (threadIdx.x == 0) r[(size_t)b * T_ + tau] = red[0] * (1.0f / (float)D_);
}

// ============ 3) top-K lags + softmax weights ============
__global__ void k_topk(const float* __restrict__ r, int* __restrict__ idx,
                       float* __restrict__ wgt) {
    if (threadIdx.x != 0) return;
    int b = blockIdx.x;
    const float* rb = r + (size_t)b * T_;
    float vals[K_]; int ids[K_];
    for (int k = 0; k < K_; ++k) {
        float best = -3.4e38f; int bi = 1;
        for (int tau = 1; tau < T_; ++tau) {          // lag 0 forbidden
            bool taken = false;
            for (int j = 0; j < k; ++j) if (ids[j] == tau) taken = true;
            if (!taken && rb[tau] > best) { best = rb[tau]; bi = tau; }
        }
        vals[k] = best; ids[k] = bi;
    }
    float m = vals[0], den = 0.f, e[K_];
    for (int k = 0; k < K_; ++k) { e[k] = expf(vals[k] - m); den += e[k]; }
    for (int k = 0; k < K_; ++k) { idx[b*K_+k] = ids[k]; wgt[b*K_+k] = e[k] / den; }
}

// ============ 4) pre-swizzle conv weights into WMMA B-fragment layout ============
// B fragment (16x16x32 f16, wave32): lane L, half j -> K = 8*(L>>4) + (j<8 ? j : j+8), N = L&15.
// frag address = ((((tap*2+ks)*4 + nt)*32 + lane)*16 + j)
__global__ void k_prep_w(const float* __restrict__ w, half_t* __restrict__ wf) {
    int i = blockIdx.x * blockDim.x + threadIdx.x;
    if (i >= 9*2*4*32*16) return;
    int j    =  i        & 15;
    int lane = (i >> 4)  & 31;
    int nt   = (i >> 9)  & 3;
    int ks   = (i >> 11) & 1;
    int tap  =  i >> 12;                       // 0..8
    int n    = nt * 16 + (lane & 15);          // out channel
    int kin  = 32 * ks + 8 * (lane >> 4) + (j < 8 ? j : j + 8); // in channel
    wf[i] = (half_t)w[((size_t)n * D_ + kin) * 9 + tap];
}

// ---- A fragment from an f16 channel row (ISA 16-bit A layout) ----
// lane L: row M = L&15;  half j -> channel c0 + (j<8 ? j : j+8), c0 = 32*ks + 8*(L>>4)
__device__ __forceinline__ v16h afrag_f16(const half_t* row, int c0) {
    v8h lo = *reinterpret_cast<const v8h*>(row + c0);
    v8h hi = *reinterpret_cast<const v8h*>(row + c0 + 16);
    v16h f;
    #pragma unroll
    for (int j = 0; j < 8; ++j) { f[j] = lo[j]; f[j+8] = hi[j]; }
    return f;
}

__device__ __forceinline__ v16h ldB(const half_t* wf, int tap, int ks, int nt, int lane) {
    return *reinterpret_cast<const v16h*>(wf + ((((size_t)tap*2 + ks)*4 + nt)*32 + lane)*16);
}

// ============ 5) conv1: h1 = gelu(bn1(conv3x3(g, w1))) over full U rows ============
__global__ __launch_bounds__(32)
void k_conv1(const half_t* __restrict__ s_h, const int* __restrict__ idx,
             const half_t* __restrict__ wf,
             const float* __restrict__ g1, const float* __restrict__ b1,
             const float* __restrict__ m1, const float* __restrict__ v1,
             half_t* __restrict__ h1) {
    int tile = blockIdx.x;          // 16-row M tile
    int bk   = blockIdx.y;          // b*K + k
    int b    = bk / K_;
    int p    = idx[bk];
    int cyc  = (T_ + p - 1) / p;
    int lane = threadIdx.x;
    int mrow = lane & 15, hi = lane >> 4;
    int u    = tile * 16 + mrow;
    int rrow = u / p, crow = u % p;
    int c0   = 8 * hi;              // + 32*ks at use
    const half_t* s_b = s_h + (size_t)b * T_ * D_;

    v8f acc[4] = {v8f{}, v8f{}, v8f{}, v8f{}};

    #pragma unroll
    for (int dr = -1; dr <= 1; ++dr) {
        #pragma unroll
        for (int dc = -1; dc <= 1; ++dc) {
            int tap = (dr + 1) * 3 + (dc + 1);
            int rr = rrow + dr, cc = crow + dc;
            bool valid = (rr >= 0) & (rr < cyc) & (cc >= 0) & (cc < p);
            int un = u + dr * p + dc;
            un = un < 0 ? 0 : (un > U_-1 ? U_-1 : un);
            int vv = un < T_ ? un : (2*T_ - 2 - un);     // reflect
            int tt = vv + p; if (tt >= T_) tt -= T_;     // roll by -p
            v16h a0 = {}, a1 = {};
            if (valid) {
                const half_t* row = s_b + (size_t)tt * D_;
                a0 = afrag_f16(row, c0);
                a1 = afrag_f16(row, c0 + 32);
            }
            #pragma unroll
            for (int nt = 0; nt < 4; ++nt) {
                acc[nt] = WMMA(a0, ldB(wf, tap, 0, nt, lane), acc[nt]);
                acc[nt] = WMMA(a1, ldB(wf, tap, 1, nt, lane), acc[nt]);
            }
        }
    }

    // epilogue: BN1 + exact GELU, store f16.  D layout: row = v + 8*hi, col = lane&15
    int ncol = lane & 15;
    #pragma unroll
    for (int nt = 0; nt < 4; ++nt) {
        int n = nt * 16 + ncol;
        float scale = g1[n] * rsqrtf(v1[n] + EPSB);
        float shift = b1[n] - m1[n] * scale;
        #pragma unroll
        for (int vreg = 0; vreg < 8; ++vreg) {
            int ur = tile * 16 + vreg + 8 * hi;
            if (ur < U_) {
                float val = acc[nt][vreg] * scale + shift;
                val = 0.5f * val * (1.0f + erff(val * 0.70710678118654752f));
                h1[((size_t)bk * UPAD + ur) * D_ + n] = (half_t)val;
            }
        }
    }
}

// ============ 6) conv2: yp = g + bn2(conv3x3(h1, w2)), crop to T ============
__global__ __launch_bounds__(32)
void k_conv2(const half_t* __restrict__ h1, const float* __restrict__ s,
             const int* __restrict__ idx, const half_t* __restrict__ wf,
             const float* __restrict__ g2, const float* __restrict__ b2,
             const float* __restrict__ m2, const float* __restrict__ v2,
             float* __restrict__ yp) {
    int tile = blockIdx.x;          // covers only t < T (128 tiles)
    int bk   = blockIdx.y;
    int b    = bk / K_;
    int p    = idx[bk];
    int cyc  = (T_ + p - 1) / p;
    int lane = threadIdx.x;
    int mrow = lane & 15, hi = lane >> 4;
    int u    = tile * 16 + mrow;
    int rrow = u / p, crow = u % p;
    int c0   = 8 * hi;
    const half_t* h_b = h1 + (size_t)bk * UPAD * D_;
    const float*  s_b = s  + (size_t)b * T_ * D_;

    v8f acc[4] = {v8f{}, v8f{}, v8f{}, v8f{}};

    #pragma unroll
    for (int dr = -1; dr <= 1; ++dr) {
        #pragma unroll
        for (int dc = -1; dc <= 1; ++dc) {
            int tap = (dr + 1) * 3 + (dc + 1);
            int rr = rrow + dr, cc = crow + dc;
            bool valid = (rr >= 0) & (rr < cyc) & (cc >= 0) & (cc < p);
            int un = u + dr * p + dc;
            un = un < 0 ? 0 : (un > U_-1 ? U_-1 : un);
            v16h a0 = {}, a1 = {};
            if (valid) {
                const half_t* row = h_b + (size_t)un * D_;
                a0 = afrag_f16(row, c0);
                a1 = afrag_f16(row, c0 + 32);
            }
            #pragma unroll
            for (int nt = 0; nt < 4; ++nt) {
                acc[nt] = WMMA(a0, ldB(wf, tap, 0, nt, lane), acc[nt]);
                acc[nt] = WMMA(a1, ldB(wf, tap, 1, nt, lane), acc[nt]);
            }
        }
    }

    int ncol = lane & 15;
    #pragma unroll
    for (int nt = 0; nt < 4; ++nt) {
        int n = nt * 16 + ncol;
        float scale = g2[n] * rsqrtf(v2[n] + EPSB);
        float shift = b2[n] - m2[n] * scale;
        #pragma unroll
        for (int vreg = 0; vreg < 8; ++vreg) {
            int t = tile * 16 + vreg + 8 * hi;        // t < 2048 by grid
            int tg = t + p; if (tg >= T_) tg -= T_;   // residual g (v(t)=t for t<T)
            float gval = s_b[(size_t)tg * D_ + n];
            float val  = acc[nt][vreg] * scale + shift;
            yp[((size_t)bk * T_ + t) * D_ + n] = gval + val;
        }
    }
}

// ============ 7) final aggregation ============
__global__ void k_final(const float* __restrict__ x, const float* __restrict__ trend,
                        const float* __restrict__ wgt, const float* __restrict__ yp,
                        float* __restrict__ out) {
    int i = blockIdx.x * blockDim.x + threadIdx.x;
    if (i >= B_*T_*D_) return;
    int d = i % D_;
    int t = (i / D_) % T_;
    int b = i / (D_ * T_);
    float acc = x[i] + trend[i];
    #pragma unroll
    for (int k = 0; k < K_; ++k) {
        acc += wgt[b*K_+k] * yp[(((size_t)(b*K_+k)) * T_ + t) * D_ + d];
    }
    out[i] = acc;
}

// ============ launch ============
extern "C" void kernel_launch(void* const* d_in, const int* in_sizes, int n_in,
                              void* d_out, int out_size, void* d_ws, size_t ws_size,
                              hipStream_t stream) {
    const float* x    = (const float*)d_in[0];
    const float* w1   = (const float*)d_in[1];
    const float* w2   = (const float*)d_in[2];
    const float* bn1g = (const float*)d_in[3];
    const float* bn1b = (const float*)d_in[4];
    const float* bn1m = (const float*)d_in[5];
    const float* bn1v = (const float*)d_in[6];
    const float* bn2g = (const float*)d_in[7];
    const float* bn2b = (const float*)d_in[8];
    const float* bn2m = (const float*)d_in[9];
    const float* bn2v = (const float*)d_in[10];

    char* ws = (char*)d_ws;
    float*  s     = (float*)(ws + OFF_S);
    float*  trend = (float*)(ws + OFF_TREND);
    float*  r     = (float*)(ws + OFF_R);
    int*    idx   = (int*)  (ws + OFF_IDX);
    float*  wgt   = (float*)(ws + OFF_WGT);
    half_t* wf1   = (half_t*)(ws + OFF_WF1);
    half_t* wf2   = (half_t*)(ws + OFF_WF2);
    half_t* h1    = (half_t*)(ws + OFF_H1);
    float*  yp    = (float*)(ws + OFF_YP);
    half_t* s_h   = (half_t*)(ws + OFF_SH);
    float*  out   = (float*)d_out;

    const int N = B_*T_*D_;

    k_trend<<<(N + 255) / 256, 256, 0, stream>>>(x, trend, s, s_h);
    k_autocorr<<<dim3(T_, B_), 256, 0, stream>>>(s, r);
    k_topk<<<B_, 32, 0, stream>>>(r, idx, wgt);
    k_prep_w<<<(9*2*4*32*16 + 255) / 256, 256, 0, stream>>>(w1, wf1);
    k_prep_w<<<(9*2*4*32*16 + 255) / 256, 256, 0, stream>>>(w2, wf2);
    k_conv1<<<dim3(UPAD / 16, B_ * K_), 32, 0, stream>>>(s_h, idx, wf1,
                                                         bn1g, bn1b, bn1m, bn1v, h1);
    k_conv2<<<dim3(T_ / 16, B_ * K_), 32, 0, stream>>>(h1, s, idx, wf2,
                                                       bn2g, bn2b, bn2m, bn2v, yp);
    k_final<<<(N + 255) / 256, 256, 0, stream>>>(x, trend, wgt, yp, out);
    (void)in_sizes; (void)n_in; (void)out_size; (void)ws_size;
}
